// GroupedAttention_38199439130911
// MI455X (gfx1250) — matrix operand order repaired and checked by reading, not compiled, not run on hardware
//
#include <hip/hip_runtime.h>
#include <cstdint>
#include <cstddef>

// ---------------- types / constants ----------------
typedef __bf16 bf16_t;
typedef __attribute__((ext_vector_type(16))) __bf16 v16bf;
typedef __attribute__((ext_vector_type(8)))  __bf16 bf16x8;
typedef __attribute__((ext_vector_type(8)))  float  v8f;
typedef __attribute__((ext_vector_type(4)))  float  f32x4;
typedef __attribute__((ext_vector_type(2)))  float  f32x2;
typedef __attribute__((__vector_size__(4 * sizeof(int)))) int v4i_t;  // builtin's b128 type

constexpr int kHQ = 32, kHK = 8, kDH = 64, kD = 2048, kB = 2, kNKV = 1024;
constexpr float kScale = 0.125f;                         // 1/sqrt(DH)
constexpr float kLog2e = 1.44269504088896340736f;
constexpr float kNegLog2Base = -13.287712379549449f;     // -log2(10000)

// ---- gfx1250 async global->LDS path (ASYNCcnt), with graceful fallback ----
#if defined(__has_builtin)
#if __has_builtin(__builtin_amdgcn_global_load_async_to_lds_b128) && \
    __has_builtin(__builtin_amdgcn_s_wait_asynccnt)
#define USE_ASYNC_LDS 1
#endif
#endif
#ifndef USE_ASYNC_LDS
#define USE_ASYNC_LDS 0
#endif

__device__ __forceinline__ void cp_b128_to_lds(bf16_t* lds_dst, const bf16_t* gsrc) {
#if USE_ASYNC_LDS
  __builtin_amdgcn_global_load_async_to_lds_b128((v4i_t*)gsrc, (v4i_t*)lds_dst, 0, 0);
#else
  *(bf16x8*)lds_dst = *(const bf16x8*)gsrc;
#endif
}
// Wait for all outstanding async copies, then workgroup barrier.
__device__ __forceinline__ void cp_wait_barrier() {
#if USE_ASYNC_LDS
  __builtin_amdgcn_s_wait_asynccnt(0);
#endif
  __syncthreads();
}

// Native f32->bf16 convert (v_cvt_pk_bf16_f32 when paired).
__device__ __forceinline__ bf16_t f2bf(float f) { return static_cast<bf16_t>(f); }

__device__ __forceinline__ v8f wmma_bf16(v16bf a, v16bf b, v8f c) {
  return __builtin_amdgcn_wmma_f32_16x16x32_bf16(false, a, false, b, (short)0, c,
                                                 false, false);
}

// A-fragment (16x32 bf16) from row-major storage, leading dim `ld`.
__device__ __forceinline__ v16bf load_a_bf16(const bf16_t* base, int ld, int lane) {
  const bf16_t* p = base + (size_t)(lane & 15) * ld + ((lane >> 4) << 3);
  bf16x8 lo = *(const bf16x8*)p;
  bf16x8 hi = *(const bf16x8*)(p + 16);
  v16bf a;
#pragma unroll
  for (int i = 0; i < 8; ++i) { a[i] = lo[i]; a[i + 8] = hi[i]; }
  return a;
}

// B-fragment (32x16 bf16) from K-contiguous storage (row n = B column n).
__device__ __forceinline__ v16bf load_b_bf16(const bf16_t* base, int ld, int lane) {
  const bf16_t* p = base + (size_t)(lane & 15) * ld + ((lane >> 4) << 3);
  bf16x8 lo = *(const bf16x8*)p;
  bf16x8 hi = *(const bf16x8*)(p + 16);
  v16bf b;
#pragma unroll
  for (int i = 0; i < 8; ++i) { b[i] = lo[i]; b[i + 8] = hi[i]; }
  return b;
}

// 16-lane-group reductions (wave32): xor masks 1,2,4,8 stay inside each half.
__device__ __forceinline__ float rmax16(float v) {
  v = fmaxf(v, __shfl_xor(v, 1, 32));
  v = fmaxf(v, __shfl_xor(v, 2, 32));
  v = fmaxf(v, __shfl_xor(v, 4, 32));
  v = fmaxf(v, __shfl_xor(v, 8, 32));
  return v;
}
__device__ __forceinline__ float rsum16(float v) {
  v += __shfl_xor(v, 1, 32);
  v += __shfl_xor(v, 2, 32);
  v += __shfl_xor(v, 4, 32);
  v += __shfl_xor(v, 8, 32);
  return v;
}

// Cooperative stage of one 16x32 f32 tile -> bf16 LDS tile (256 threads, 2 elems each).
__device__ __forceinline__ void stage_a_f32(const float* A, int k0, bf16_t* buf, int tid) {
  const int e = tid << 1;
  const int row = e >> 5, col = e & 31;
  f32x2 v = *(const f32x2*)(A + (size_t)row * kD + k0 + col);
  union { bf16_t h[2]; uint32_t u; } pk;
  pk.h[0] = f2bf(v[0]); pk.h[1] = f2bf(v[1]);
  *(uint32_t*)(buf + e) = pk.u;
}

// ---------------- kernel 1: weight fp32 -> bf16, transposed to [N][K] ----------------
__global__ __launch_bounds__(256) void k_w_transpose(const float* __restrict__ src,
                                                     bf16_t* __restrict__ dst,
                                                     int K, int N) {
  int idx = blockIdx.x * 256 + threadIdx.x;     // over (K/2)*N
  if (idx >= (K >> 1) * N) return;
  int kp = idx / N, n = idx - kp * N;
  int k = kp << 1;
  union { bf16_t h[2]; uint32_t u; } pack;
  pack.h[0] = f2bf(src[(size_t)k * N + n]);
  pack.h[1] = f2bf(src[(size_t)(k + 1) * N + n]);
  *(uint32_t*)(dst + (size_t)n * K + k) = pack.u;
}

// ---------------- kernel 2: Q projection + fused RoPE ----------------
// x-tile staged once per block into LDS (converted), shared by all 8 waves.
__global__ __launch_bounds__(256) void k_proj_q(const float* __restrict__ x,
                                                const bf16_t* __restrict__ WqT,
                                                bf16_t* __restrict__ Q,
                                                int T, int tshift) {
  __shared__ __align__(16) bf16_t Abuf[2][16 * 32];
  const int tid = threadIdx.x;
  const int lane = tid & 31, w = tid >> 5;
  const int ln = lane & 15, rg = lane >> 4;
  const int m0 = blockIdx.x << 4;
  const int h = blockIdx.y * 4 + (w >> 1);
  const int dh0 = (w & 1) << 4;
  const float* A = x + (size_t)m0 * kD;
  const bf16_t* B0 = WqT + (size_t)(h * kDH + dh0) * kD;
  const bf16_t* B1 = WqT + (size_t)(h * kDH + dh0 + 32) * kD;
  v8f c0 = {}, c1 = {};
  stage_a_f32(A, 0, Abuf[0], tid);
  for (int k0 = 0; k0 < kD; k0 += 32) {
    __syncthreads();
    const int cur = (k0 >> 5) & 1;
    if (k0 + 32 < kD) stage_a_f32(A, k0 + 32, Abuf[cur ^ 1], tid);
    __builtin_prefetch(B0 + k0 + 64 + (size_t)ln * kD, 0, 1);
    v16bf a = load_a_bf16(Abuf[cur], 32, lane);
    c0 = wmma_bf16(a, load_b_bf16(B0 + k0, kD, lane), c0);
    c1 = wmma_bf16(a, load_b_bf16(B1 + k0, kD, lane), c1);
  }
  const float inv = __builtin_amdgcn_exp2f(kNegLog2Base * (float)(dh0 + ln) * (1.0f / 32.0f));
#pragma unroll
  for (int r = 0; r < 8; ++r) {
    const int m = m0 + r + (rg << 3);
    const int bi = m >> tshift, t = m & (T - 1);
    const float ang = (float)t * inv;
    const float cs = __cosf(ang), sn = __sinf(ang);
    const float lo = c0[r], hi = c1[r];
    bf16_t* dst = Q + (((size_t)bi * kHQ + h) * T + t) * kDH + dh0 + ln;
    dst[0]  = f2bf(lo * cs - hi * sn);   // dh < 32 half
    dst[32] = f2bf(hi * cs + lo * sn);   // dh >= 32 half
  }
}

// ---------------- kernel 3: KV projection; K gets RoPE, V stored transposed ----------------
__global__ __launch_bounds__(256) void k_proj_kv(const float* __restrict__ x,
                                                 const bf16_t* __restrict__ WkvT,
                                                 bf16_t* __restrict__ Kc,
                                                 bf16_t* __restrict__ Vt,
                                                 int T, int tshift) {
  __shared__ __align__(16) bf16_t Abuf[2][16 * 32];
  const int tid = threadIdx.x;
  const int lane = tid & 31, w = tid >> 5;
  const int ln = lane & 15, rg = lane >> 4;
  const int m0 = blockIdx.x << 4;
  const int hk = blockIdx.y * 2 + (w >> 2);
  const int role = w & 3;                    // 0,1: K (RoPE pairs); 2,3: V
  int col0, col1;
  if (role < 2) { col0 = hk * kDH + (role << 4);                    col1 = col0 + 32; }
  else          { col0 = kHK * kDH + hk * kDH + ((role - 2) << 5);  col1 = col0 + 16; }
  const float* A = x + (size_t)m0 * kD;
  const bf16_t* B0 = WkvT + (size_t)col0 * kD;
  const bf16_t* B1 = WkvT + (size_t)col1 * kD;
  v8f c0 = {}, c1 = {};
  stage_a_f32(A, 0, Abuf[0], tid);
  for (int k0 = 0; k0 < kD; k0 += 32) {
    __syncthreads();
    const int cur = (k0 >> 5) & 1;
    if (k0 + 32 < kD) stage_a_f32(A, k0 + 32, Abuf[cur ^ 1], tid);
    __builtin_prefetch(B0 + k0 + 64 + (size_t)ln * kD, 0, 1);
    v16bf a = load_a_bf16(Abuf[cur], 32, lane);
    c0 = wmma_bf16(a, load_b_bf16(B0 + k0, kD, lane), c0);
    c1 = wmma_bf16(a, load_b_bf16(B1 + k0, kD, lane), c1);
  }
  if (role < 2) {
    const int dh0 = role << 4;
    const float inv = __builtin_amdgcn_exp2f(kNegLog2Base * (float)(dh0 + ln) * (1.0f / 32.0f));
#pragma unroll
    for (int r = 0; r < 8; ++r) {
      const int m = m0 + r + (rg << 3);
      const int bi = m >> tshift, t = m & (T - 1);
      const float ang = (float)t * inv;
      const float cs = __cosf(ang), sn = __sinf(ang);
      const float lo = c0[r], hi = c1[r];
      bf16_t* dst = Kc + (((size_t)bi * kHK + hk) * T + t) * kDH + dh0 + ln;
      dst[0]  = f2bf(lo * cs - hi * sn);
      dst[32] = f2bf(hi * cs + lo * sn);
    }
  } else {
    const int dh0 = (role - 2) << 5;
#pragma unroll
    for (int r = 0; r < 8; ++r) {
      const int m = m0 + r + (rg << 3);
      const int bi = m >> tshift, t = m & (T - 1);
      bf16_t* dst = Vt + (((size_t)bi * kHK + hk) * kDH + dh0 + ln) * (size_t)T + t;
      dst[0]              = f2bf(c0[r]);
      dst[(size_t)16 * T] = f2bf(c1[r]);
    }
  }
}

// ---------------- kernel 4: causal flash attention ----------------
// One 4-wave block per (batch, kv-head, 16-row q tile): the 4 q heads of a
// GQA group share double-buffered K/V tiles staged into LDS via the async
// global->LDS DMA path (ASYNCcnt), overlapping the next tile's copy with WMMA.
__global__ __launch_bounds__(128) void k_attn(const bf16_t* __restrict__ Q,
                                              const bf16_t* __restrict__ Kc,
                                              const bf16_t* __restrict__ Vt,
                                              bf16_t* __restrict__ AO,
                                              int T, int tshift) {
  __shared__ __align__(16) bf16_t Kbuf[2][32 * kDH];   // [t_rel][dh], 4KB each
  __shared__ __align__(16) bf16_t Vbuf[2][kDH * 32];   // [dh][t_rel], 4KB each
  __shared__ __align__(16) bf16_t Pst[4][16 * 32];     // per-wave P staging
  const int tid = threadIdx.x;
  const int lane = tid & 31, w = tid >> 5;
  const int ln = lane & 15, rg = lane >> 4;

  const int nqLog = tshift - 4;                 // log2(T/16)
  const int qt = blockIdx.x & ((1 << nqLog) - 1);
  const int tmp = blockIdx.x >> nqLog;
  const int hk = tmp & (kHK - 1);
  const int bi = tmp >> 3;
  const int h = (hk << 2) + w;                  // this wave's q head
  const int q0 = qt << 4;

  const bf16_t* Qh = Q  + (((size_t)bi * kHQ + h)  * T) * kDH;
  const bf16_t* Kh = Kc + (((size_t)bi * kHK + hk) * T) * kDH;
  const bf16_t* Vh = Vt + (((size_t)bi * kHK + hk) * kDH) * T;

  v16bf aq0 = load_a_bf16(Qh + (size_t)q0 * kDH, kDH, lane);        // dh 0..31
  v16bf aq1 = load_a_bf16(Qh + (size_t)q0 * kDH + 32, kDH, lane);   // dh 32..63

  v8f acc0 = {}, acc1 = {}, acc2 = {}, acc3 = {};
  float m_r[8], l_r[8];
#pragma unroll
  for (int r = 0; r < 8; ++r) { m_r[r] = -1e30f; l_r[r] = 0.f; }

  const int nblk = (q0 + 16 + 31) >> 5;         // 32 keys per iteration

  // stage block 0
#pragma unroll
  for (int c = tid; c < 256; c += 128) {        // K tile: 32x64 bf16
    const int e = c << 3;
    cp_b128_to_lds(&Kbuf[0][e], Kh + (size_t)(e >> 6) * kDH + (e & 63));
  }
#pragma unroll
  for (int c = tid; c < 256; c += 128) {        // V tile: 64x32 bf16
    const int e = c << 3;
    cp_b128_to_lds(&Vbuf[0][e], Vh + (size_t)(e >> 5) * T + (e & 31));
  }

  for (int j = 0; j < nblk; ++j) {
    const int k0 = j << 5;
    cp_wait_barrier();                          // stage j complete, visible to all waves
    const int cur = j & 1;
    if (j + 1 < nblk) {                         // kick off stage j+1 into other buffer
      const int kn = k0 + 32;
#pragma unroll
      for (int c = tid; c < 256; c += 128) {
        const int e = c << 3;
        cp_b128_to_lds(&Kbuf[cur ^ 1][e], Kh + (size_t)(kn + (e >> 6)) * kDH + (e & 63));
      }
#pragma unroll
      for (int c = tid; c < 256; c += 128) {
        const int e = c << 3;
        cp_b128_to_lds(&Vbuf[cur ^ 1][e], Vh + (size_t)(e >> 5) * T + kn + (e & 31));
      }
    }
    const bf16_t* kb = Kbuf[cur];
    const bf16_t* vb = Vbuf[cur];
    v8f s0 = {}, s1 = {};
    s0 = wmma_bf16(aq0, load_b_bf16(kb,            kDH, lane), s0);
    s0 = wmma_bf16(aq1, load_b_bf16(kb + 32,       kDH, lane), s0);
    s1 = wmma_bf16(aq0, load_b_bf16(kb + 16 * kDH,      kDH, lane), s1);
    s1 = wmma_bf16(aq1, load_b_bf16(kb + 16 * kDH + 32, kDH, lane), s1);
#pragma unroll
    for (int r = 0; r < 8; ++r) {
      const int row = q0 + r + (rg << 3);
      const int cA = k0 + ln, cB = cA + 16;
      float f0 = (cA > row) ? -1e30f : s0[r] * kScale;
      float f1 = (cB > row) ? -1e30f : s1[r] * kScale;
      const float mx = rmax16(fmaxf(f0, f1));
      const float m_new = fmaxf(m_r[r], mx);
      const float sc = __builtin_amdgcn_exp2f((m_r[r] - m_new) * kLog2e);
      const float p0 = __builtin_amdgcn_exp2f((f0 - m_new) * kLog2e);
      const float p1 = __builtin_amdgcn_exp2f((f1 - m_new) * kLog2e);
      l_r[r] = l_r[r] * sc + rsum16(p0 + p1);
      m_r[r] = m_new;
      acc0[r] *= sc; acc1[r] *= sc; acc2[r] *= sc; acc3[r] *= sc;
      const int prow = r + (rg << 3);
      Pst[w][prow * 32 + ln]      = f2bf(p0);
      Pst[w][prow * 32 + 16 + ln] = f2bf(p1);
    }
    __syncthreads();                            // P visible (and K/V reads ordered)
    v16bf pa = load_a_bf16(Pst[w], 32, lane);   // P as 16x32 A-fragment
    acc0 = wmma_bf16(pa, load_b_bf16(vb,           32, lane), acc0);
    acc1 = wmma_bf16(pa, load_b_bf16(vb + 16 * 32, 32, lane), acc1);
    acc2 = wmma_bf16(pa, load_b_bf16(vb + 32 * 32, 32, lane), acc2);
    acc3 = wmma_bf16(pa, load_b_bf16(vb + 48 * 32, 32, lane), acc3);
  }
#pragma unroll
  for (int r = 0; r < 8; ++r) {
    const int trow = q0 + r + (rg << 3);
    const float rl = __builtin_amdgcn_rcpf(l_r[r]);
    bf16_t* dst = AO + ((size_t)bi * T + trow) * (size_t)(kHQ * kDH) + h * kDH + ln;
    dst[0]  = f2bf(acc0[r] * rl);
    dst[16] = f2bf(acc1[r] * rl);
    dst[32] = f2bf(acc2[r] * rl);
    dst[48] = f2bf(acc3[r] * rl);
  }
}

// ---------------- kernel 5: output projection (bf16 x bf16 -> f32 out) ----------------
// A-tile (shared by all 8 waves) staged via async global->LDS copies.
__global__ __launch_bounds__(256) void k_out_proj(const bf16_t* __restrict__ AO,
                                                  const bf16_t* __restrict__ WoT,
                                                  float* __restrict__ out, int T) {
  __shared__ __align__(16) bf16_t Abuf[2][16 * 32];    // 1KB each
  const int tid = threadIdx.x;
  const int lane = tid & 31, w = tid >> 5;
  const int ln = lane & 15, rg = lane >> 4;
  const int m0 = blockIdx.x << 4;
  const int n0 = blockIdx.y * 256 + w * 32;
  const bf16_t* A  = AO  + (size_t)m0 * kD;
  const bf16_t* B0 = WoT + (size_t)n0 * kD;
  const bf16_t* B1 = WoT + (size_t)(n0 + 16) * kD;
  v8f c0 = {}, c1 = {};
  if (tid < 64) {                               // 64 x b128 = 1KB tile
    const int e = tid << 3;
    cp_b128_to_lds(&Abuf[0][e], A + (size_t)(e >> 5) * kD + (e & 31));
  }
  for (int k0 = 0; k0 < kD; k0 += 32) {
    cp_wait_barrier();
    const int cur = (k0 >> 5) & 1;
    if (k0 + 32 < kD && tid < 64) {
      const int e = tid << 3;
      cp_b128_to_lds(&Abuf[cur ^ 1][e], A + (size_t)(e >> 5) * kD + k0 + 32 + (e & 31));
    }
    __builtin_prefetch(B0 + k0 + 64 + (size_t)ln * kD, 0, 1);
    v16bf a = load_a_bf16(Abuf[cur], 32, lane);
    c0 = wmma_bf16(a, load_b_bf16(B0 + k0, kD, lane), c0);
    c1 = wmma_bf16(a, load_b_bf16(B1 + k0, kD, lane), c1);
  }
#pragma unroll
  for (int r = 0; r < 8; ++r) {
    const int m = m0 + r + (rg << 3);
    out[(size_t)m * kD + n0 + ln]      = c0[r];
    out[(size_t)m * kD + n0 + 16 + ln] = c1[r];
  }
}

// ---------------- host launch ----------------
extern "C" void kernel_launch(void* const* d_in, const int* in_sizes, int n_in,
                              void* d_out, int out_size, void* d_ws, size_t ws_size,
                              hipStream_t stream) {
  const float* x   = (const float*)d_in[0];
  const float* Wq  = (const float*)d_in[1];
  const float* Wkv = (const float*)d_in[2];
  const float* Wo  = (const float*)d_in[3];
  float* out = (float*)d_out;
  const int T = in_sizes[0] / (kB * kD);
  const int BT = kB * T;
  int tshift = 0;
  while ((1 << tshift) < T) ++tshift;           // T is a power of two (2048)
  (void)n_in; (void)out_size; (void)ws_size;

  char* p = (char*)d_ws;                        // ~60 MB total
  bf16_t* WqT  = (bf16_t*)p; p += (size_t)kD * kD   * sizeof(bf16_t);
  bf16_t* WkvT = (bf16_t*)p; p += (size_t)kNKV * kD * sizeof(bf16_t);
  bf16_t* WoT  = (bf16_t*)p; p += (size_t)kD * kD   * sizeof(bf16_t);
  bf16_t* Qb   = (bf16_t*)p; p += (size_t)BT * kHQ * kDH * sizeof(bf16_t);
  bf16_t* Kb   = (bf16_t*)p; p += (size_t)BT * kHK * kDH * sizeof(bf16_t);
  bf16_t* Vtb  = (bf16_t*)p; p += (size_t)BT * kHK * kDH * sizeof(bf16_t);
  bf16_t* AOb  = (bf16_t*)p; p += (size_t)BT * kHQ * kDH * sizeof(bf16_t);

  k_w_transpose<<<((kD / 2) * kD   + 255) / 256, 256, 0, stream>>>(Wq,  WqT,  kD, kD);
  k_w_transpose<<<((kD / 2) * kNKV + 255) / 256, 256, 0, stream>>>(Wkv, WkvT, kD, kNKV);
  k_w_transpose<<<((kD / 2) * kD   + 255) / 256, 256, 0, stream>>>(Wo,  WoT,  kD, kD);

  dim3 gq(BT / 16, kHQ / 4);
  k_proj_q<<<gq, 256, 0, stream>>>(x, WqT, Qb, T, tshift);
  dim3 gkv(BT / 16, kHK / 2);
  k_proj_kv<<<gkv, 256, 0, stream>>>(x, WkvT, Kb, Vtb, T, tshift);

  k_attn<<<kB * kHK * (T / 16), 128, 0, stream>>>(Qb, Kb, Vtb, AOb, T, tshift);

  dim3 go(BT / 16, kD / 256);
  k_out_proj<<<go, 256, 0, stream>>>(AOb, WoT, out, T);
}
